// MultiHeadAttn_32693291057795
// MI455X (gfx1250) — matrix operand order repaired and checked by reading, not compiled, not run on hardware
//
#include <hip/hip_runtime.h>
#include <hip/hip_bf16.h>
#include <cstdint>
#include <cstddef>

// ---------------------------------------------------------------------------
// Types for CDNA5 WMMA (wave32): A/B fragments are v16 bf16, C/D are v8 f32.
// ---------------------------------------------------------------------------
typedef __bf16 bf16_t;
typedef __attribute__((ext_vector_type(16))) __bf16 v16bf;
typedef __attribute__((ext_vector_type(8)))  __bf16 v8bf;
typedef __attribute__((ext_vector_type(8)))  float  v8f;

union ABFrag { v16bf v; v8bf h[2]; };

__device__ __forceinline__ v8bf ld8(const bf16_t* p) {
    return *reinterpret_cast<const v8bf*>(p);
}

#define BATCH 2
#define SEQ   2048
#define DE    1024
#define NH    16
#define DQ    64
#define M_TOTAL (BATCH * SEQ)   // 4096 rows

// ---------------------------------------------------------------------------
// fp32 -> bf16 conversion kernels
// ---------------------------------------------------------------------------
__global__ void cvt_bf16_kernel(const float* __restrict__ in,
                                bf16_t* __restrict__ out, int n) {
    int i = blockIdx.x * blockDim.x + threadIdx.x;
    if (i < n) out[i] = (bf16_t)in[i];
}

// in: (DE,DE) row-major W[k][n]; out: Wt[n][k] (so WMMA B-fragments read
// two runs of 8 contiguous K values per lane).
__global__ void cvt_transpose_kernel(const float* __restrict__ in,
                                     bf16_t* __restrict__ out) {
    int i = blockIdx.x * blockDim.x + threadIdx.x;
    int n = i >> 10, k = i & 1023;
    out[((size_t)n << 10) + k] = (bf16_t)in[((size_t)k << 10) + n];
}

// ---------------------------------------------------------------------------
// WMMA bf16 GEMM: C[M,N] = A[M,K] * Bt[N,K]^T + bias
//   8 waves/block, each wave computes a 16x64 tile (4 N-subtiles), K step 32.
//   mode 0: out bf16 row-major, value scaled by `scale` (Q gets 1/sqrt(64))
//   mode 1: out bf16 transposed per batch: out[b][n][row%SEQ]  (V^T for attn)
//   mode 2: out fp32 = acc + bias + resid  (output projection + residual)
// ---------------------------------------------------------------------------
__global__ __launch_bounds__(256)
void gemm_bf16_kernel(const bf16_t* __restrict__ A, const bf16_t* __restrict__ Bt,
                      const float* __restrict__ bias,
                      bf16_t* __restrict__ outb, float* __restrict__ outf,
                      const float* __restrict__ resid,
                      int M, int N, int K, int mode, float scale)
{
    const int lane = threadIdx.x & 31;
    const int wv   = threadIdx.x >> 5;
    const int l16  = lane & 15;
    const int g    = lane >> 4;            // half-wave group
    const int m0   = blockIdx.y * 128 + wv * 16;
    const int n0   = blockIdx.x * 64;

    const bf16_t* ap = A + (size_t)(m0 + l16) * K;
    const bf16_t* bp[4];
#pragma unroll
    for (int t = 0; t < 4; ++t)
        bp[t] = Bt + (size_t)(n0 + t * 16 + l16) * K;

    v8f acc[4] = {};

    for (int kk = 0; kk < K; kk += 32) {
        // A fragment 16x32: lane row = l16; elems 0..7 -> K = kk + g*8 + e,
        // elems 8..15 -> K = kk + 16 + g*8 + e   (ISA 7.12.2, 16-bit A)
        ABFrag a;
        a.h[0] = ld8(ap + kk + g * 8);
        a.h[1] = ld8(ap + kk + 16 + g * 8);
        __builtin_prefetch(ap + kk + 512, 0, 1);  // global_prefetch_b8
#pragma unroll
        for (int t = 0; t < 4; ++t) {
            // B fragment 32x16: lane col = l16; K = kk + g*8 + e, kk + 16 + g*8 + e
            ABFrag b;
            b.h[0] = ld8(bp[t] + kk + g * 8);
            b.h[1] = ld8(bp[t] + kk + 16 + g * 8);
            acc[t] = __builtin_amdgcn_wmma_f32_16x16x32_bf16(
                false, a.v, false, b.v, (short)0, acc[t], false, false);
        }
    }

    if (mode == 0) {
#pragma unroll
        for (int t = 0; t < 4; ++t) {
            int col = n0 + t * 16 + l16;
            float bi = bias[col];
#pragma unroll
            for (int r = 0; r < 8; ++r) {
                int row = m0 + r + 8 * g;   // C layout: VGPR r -> row r + 8*g
                outb[(size_t)row * N + col] = (bf16_t)((acc[t][r] + bi) * scale);
            }
        }
    } else if (mode == 1) {
        const int bidx = m0 / SEQ;               // tiles never straddle batches
        const int key0 = (m0 % SEQ) + 8 * g;     // rows r=0..7 are contiguous keys
#pragma unroll
        for (int t = 0; t < 4; ++t) {
            int col = n0 + t * 16 + l16;
            float bi = bias[col];
            v8bf pk;
#pragma unroll
            for (int r = 0; r < 8; ++r) pk[r] = (bf16_t)(acc[t][r] + bi);
            *reinterpret_cast<v8bf*>(outb + (size_t)bidx * N * SEQ
                                          + (size_t)col * SEQ + key0) = pk;
        }
    } else {
#pragma unroll
        for (int t = 0; t < 4; ++t) {
            int col = n0 + t * 16 + l16;
            float bi = bias[col];
#pragma unroll
            for (int r = 0; r < 8; ++r) {
                int row = m0 + r + 8 * g;
                size_t idx = (size_t)row * N + col;
                outf[idx] = acc[t][r] + bi + resid[idx];
            }
        }
    }
}

// ---------------------------------------------------------------------------
// Flash attention: one wave per 16-query tile of one (b,h).
//   Q is pre-scaled by 1/sqrt(DQ). K row-major (b,key,h*64+d). V transposed
//   (b, d, key) so P@V B-fragments read contiguous keys.
//   8 WMMAs per 32-key step; online softmax via __shfl_xor over 16-lane groups.
// ---------------------------------------------------------------------------
__global__ __launch_bounds__(256)
void attn_kernel(const bf16_t* __restrict__ Q, const bf16_t* __restrict__ Km,
                 const bf16_t* __restrict__ Vt, bf16_t* __restrict__ ctx)
{
    __shared__ alignas(16) bf16_t pbuf[8][16][32];   // per-wave P staging

    const int lane = threadIdx.x & 31;
    const int wv   = threadIdx.x >> 5;
    const int l16  = lane & 15;
    const int g    = lane >> 4;

    const int wid = blockIdx.x * 8 + wv;             // 4096 waves total
    const int qt  = wid & (SEQ / 16 - 1);            // 128 q-tiles
    const int h   = (wid >> 7) & (NH - 1);
    const int b   = wid >> 11;
    const int q0  = qt * 16;

    // Q fragments for kd 0..63 (two 16x32 A fragments), loaded once
    const bf16_t* qp = Q + ((size_t)(b * SEQ + q0 + l16) << 10) + h * DQ;
    ABFrag qf0, qf1;
    qf0.h[0] = ld8(qp + g * 8);        qf0.h[1] = ld8(qp + 16 + g * 8);
    qf1.h[0] = ld8(qp + 32 + g * 8);   qf1.h[1] = ld8(qp + 48 + g * 8);

    float mrow[8], lrow[8];
    v8f acc[4] = {};
#pragma unroll
    for (int r = 0; r < 8; ++r) { mrow[r] = -1e30f; lrow[r] = 0.f; }

    const bf16_t* kbase = Km + ((size_t)(b * SEQ) << 10) + h * DQ;
    const bf16_t* vbase = Vt + ((size_t)b * DE + h * DQ) * SEQ;

    for (int j0 = 0; j0 < SEQ; j0 += 32) {
        // ---- scores: two 16x16 tiles over 32 keys, kdim 64 = 2 WMMA steps
        v8f s0 = {}, s1 = {};
        const bf16_t* kp0 = kbase + ((size_t)(j0 + l16) << 10);
        const bf16_t* kp1 = kbase + ((size_t)(j0 + 16 + l16) << 10);
#pragma unroll
        for (int kk = 0; kk < 64; kk += 32) {
            ABFrag k0, k1;
            k0.h[0] = ld8(kp0 + kk + g * 8); k0.h[1] = ld8(kp0 + kk + 16 + g * 8);
            k1.h[0] = ld8(kp1 + kk + g * 8); k1.h[1] = ld8(kp1 + kk + 16 + g * 8);
            const v16bf qv = (kk == 0) ? qf0.v : qf1.v;
            s0 = __builtin_amdgcn_wmma_f32_16x16x32_bf16(
                false, qv, false, k0.v, (short)0, s0, false, false);
            s1 = __builtin_amdgcn_wmma_f32_16x16x32_bf16(
                false, qv, false, k1.v, (short)0, s1, false, false);
        }

        // ---- online softmax (C layout: VGPR r -> row r + 8*g, col = l16)
#pragma unroll
        for (int r = 0; r < 8; ++r) {
            float t = fmaxf(s0[r], s1[r]);
            t = fmaxf(t, __shfl_xor(t, 1, 32));
            t = fmaxf(t, __shfl_xor(t, 2, 32));
            t = fmaxf(t, __shfl_xor(t, 4, 32));
            t = fmaxf(t, __shfl_xor(t, 8, 32));
            float mn   = fmaxf(mrow[r], t);
            float corr = __expf(mrow[r] - mn);
            float p0 = __expf(s0[r] - mn);
            float p1 = __expf(s1[r] - mn);
            s0[r] = p0; s1[r] = p1;
            float rs = p0 + p1;
            rs += __shfl_xor(rs, 1, 32);
            rs += __shfl_xor(rs, 2, 32);
            rs += __shfl_xor(rs, 4, 32);
            rs += __shfl_xor(rs, 8, 32);
            lrow[r] = lrow[r] * corr + rs;
            mrow[r] = mn;
#pragma unroll
            for (int t2 = 0; t2 < 4; ++t2) acc[t2][r] *= corr;
        }

        // ---- relayout P (C layout) -> A fragment via LDS
#pragma unroll
        for (int r = 0; r < 8; ++r) {
            pbuf[wv][r + 8 * g][l16]      = (bf16_t)s0[r];
            pbuf[wv][r + 8 * g][16 + l16] = (bf16_t)s1[r];
        }
        asm volatile("s_wait_dscnt 0" ::: "memory");   // wave-local LDS RAW
        ABFrag pf;
        pf.h[0] = *reinterpret_cast<const v8bf*>(&pbuf[wv][l16][g * 8]);
        pf.h[1] = *reinterpret_cast<const v8bf*>(&pbuf[wv][l16][16 + g * 8]);

        // ---- ctx += P(16x32) @ V(32x64): 4 N-subtiles, single k-step
#pragma unroll
        for (int nt = 0; nt < 4; ++nt) {
            const bf16_t* vp = vbase + (size_t)(nt * 16 + l16) * SEQ + j0 + g * 8;
            ABFrag vf;
            vf.h[0] = ld8(vp);
            vf.h[1] = ld8(vp + 16);
            acc[nt] = __builtin_amdgcn_wmma_f32_16x16x32_bf16(
                false, pf.v, false, vf.v, (short)0, acc[nt], false, false);
        }
    }

    // ---- normalize and store ctx as (b, q, h*64+d) bf16
#pragma unroll
    for (int r = 0; r < 8; ++r) {
        float inv = 1.0f / lrow[r];
        int qrow = q0 + r + 8 * g;
        bf16_t* cp = ctx + ((size_t)(b * SEQ + qrow) << 10) + h * DQ;
#pragma unroll
        for (int nt = 0; nt < 4; ++nt)
            cp[nt * 16 + l16] = (bf16_t)(acc[nt][r] * inv);
    }
}

// ---------------------------------------------------------------------------
// Row LayerNorm over DE=1024, one block (256 threads) per row.
// ---------------------------------------------------------------------------
__global__ __launch_bounds__(256)
void layernorm_kernel(const float* __restrict__ x, const float* __restrict__ gamma,
                      const float* __restrict__ beta, float* __restrict__ out)
{
    __shared__ float ssum[256];
    __shared__ float ssq[256];
    const int row = blockIdx.x;
    const int tid = threadIdx.x;
    const float* xr = x + ((size_t)row << 10);
    float v[4];
    float s = 0.f, s2 = 0.f;
#pragma unroll
    for (int k = 0; k < 4; ++k) {
        v[k] = xr[tid + 256 * k];
        s += v[k]; s2 += v[k] * v[k];
    }
    ssum[tid] = s; ssq[tid] = s2;
    __syncthreads();
    for (int off = 128; off > 0; off >>= 1) {
        if (tid < off) { ssum[tid] += ssum[tid + off]; ssq[tid] += ssq[tid + off]; }
        __syncthreads();
    }
    float mu   = ssum[0] * (1.0f / 1024.0f);
    float var  = ssq[0] * (1.0f / 1024.0f) - mu * mu;
    float rstd = rsqrtf(var + 1e-5f);
    float* orow = out + ((size_t)row << 10);
#pragma unroll
    for (int k = 0; k < 4; ++k) {
        int c = tid + 256 * k;
        orow[c] = (v[k] - mu) * rstd * gamma[c] + beta[c];
    }
}

// ---------------------------------------------------------------------------
// Launch
// ---------------------------------------------------------------------------
extern "C" void kernel_launch(void* const* d_in, const int* in_sizes, int n_in,
                              void* d_out, int out_size, void* d_ws, size_t ws_size,
                              hipStream_t stream)
{
    const float* pre_q = (const float*)d_in[0];
    const float* pre_k = (const float*)d_in[1];
    const float* Wq    = (const float*)d_in[2];
    const float* bq    = (const float*)d_in[3];
    const float* Wk    = (const float*)d_in[4];
    const float* bk    = (const float*)d_in[5];
    const float* Wv    = (const float*)d_in[6];
    const float* bv    = (const float*)d_in[7];
    const float* Wo    = (const float*)d_in[8];
    const float* bo    = (const float*)d_in[9];
    const float* gamma = (const float*)d_in[10];
    const float* beta  = (const float*)d_in[11];
    float* out = (float*)d_out;

    char* ws = (char*)d_ws;
    const size_t MB = 1ull << 20;
    bf16_t* preq16 = (bf16_t*)(ws + 0 * MB);
    bf16_t* prek16 = (bf16_t*)(ws + 8 * MB);
    bf16_t* Wqt    = (bf16_t*)(ws + 16 * MB);
    bf16_t* Wkt    = (bf16_t*)(ws + 18 * MB);
    bf16_t* Wvt    = (bf16_t*)(ws + 20 * MB);
    bf16_t* Wot    = (bf16_t*)(ws + 22 * MB);
    bf16_t* Q16    = (bf16_t*)(ws + 24 * MB);
    bf16_t* K16    = (bf16_t*)(ws + 32 * MB);
    bf16_t* Vt16   = (bf16_t*)(ws + 40 * MB);
    bf16_t* ctx16  = (bf16_t*)(ws + 48 * MB);
    float*  xbuf   = (float*) (ws + 56 * MB);

    const int nElem = M_TOTAL * DE;   // 4,194,304
    cvt_bf16_kernel<<<nElem / 256, 256, 0, stream>>>(pre_q, preq16, nElem);
    cvt_bf16_kernel<<<nElem / 256, 256, 0, stream>>>(pre_k, prek16, nElem);
    cvt_transpose_kernel<<<(DE * DE) / 256, 256, 0, stream>>>(Wq, Wqt);
    cvt_transpose_kernel<<<(DE * DE) / 256, 256, 0, stream>>>(Wk, Wkt);
    cvt_transpose_kernel<<<(DE * DE) / 256, 256, 0, stream>>>(Wv, Wvt);
    cvt_transpose_kernel<<<(DE * DE) / 256, 256, 0, stream>>>(Wo, Wot);

    dim3 ggrid(DE / 64, M_TOTAL / 128);   // (16, 32)
    // Q pre-scaled by 1/sqrt(DQ) = 0.125
    gemm_bf16_kernel<<<ggrid, 256, 0, stream>>>(preq16, Wqt, bq, Q16, nullptr,
                                                nullptr, M_TOTAL, DE, DE, 0, 0.125f);
    gemm_bf16_kernel<<<ggrid, 256, 0, stream>>>(prek16, Wkt, bk, K16, nullptr,
                                                nullptr, M_TOTAL, DE, DE, 0, 1.0f);
    gemm_bf16_kernel<<<ggrid, 256, 0, stream>>>(prek16, Wvt, bv, Vt16, nullptr,
                                                nullptr, M_TOTAL, DE, DE, 1, 1.0f);

    attn_kernel<<<(BATCH * NH * (SEQ / 16)) / 8, 256, 0, stream>>>(Q16, K16, Vt16, ctx16);

    gemm_bf16_kernel<<<ggrid, 256, 0, stream>>>(ctx16, Wot, bo, nullptr, xbuf,
                                                pre_q, M_TOTAL, DE, DE, 2, 1.0f);

    layernorm_kernel<<<M_TOTAL, 256, 0, stream>>>(xbuf, gamma, beta, out);
}
